// RWKV7_78443282694938
// MI455X (gfx1250) — compile-verified, matrix operand order checked by reading, not run on hardware
//
#include <hip/hip_runtime.h>

// ---------------------------------------------------------------------------
// RWKV7 TTS forward for gfx1250 (MI455X).
// All GEMMs: bf16 WMMA (v_wmma_f32_16x16x32_bf16), f32 accumulate.
// Weights are cast+transposed to bf16 [N,K] once per launch so every WMMA
// fragment (A and B) is two contiguous 16-byte loads. Each wave computes a
// 32x64 output tile = 8 WMMAs per K-step; GEMMs with N % 128 == 0 use a
// branch-free inner loop (no exec-mask juggling around the WMMAs).
// Params assumed flattened in JAX pytree order (dict keys sorted; capitals
// sort before lowercase):
//   d_in[0]=text  d_in[1]=mask(unused)  d_in[2]=audio
//   d_in[3]=audio_embed
//   d_in[4..36]   = block0: att{Wk,Wo,Wr,Wv,a0,a1,a2,g1,g2,k_a,k_k,ln_x_b,
//                   ln_x_w,r_k,v0,v1,v2,w0,w1,w2,x_a,x_g,x_k,x_r,x_v,x_w},
//                   ffn{Wk,Wv,x_k}, ln1_b, ln1_w, ln2_b, ln2_w
//   d_in[37..69]  = block1 (same layout)
//   d_in[70]=head  71=ln_in_b 72=ln_in_w 73=ln_out_b 74=ln_out_w 75=text_embed
// ---------------------------------------------------------------------------

typedef __attribute__((ext_vector_type(16))) __bf16 v16bf;
typedef __attribute__((ext_vector_type(8)))  __bf16 v8bf;
typedef __attribute__((ext_vector_type(8)))  float  v8f;

#define C_DIM   1024
#define HEADS   16
#define HEAD_N  64

// ---- weight cast + transpose: in f32 [K,N] -> out bf16 [N,K] ---------------
__global__ void cast_transpose(const float* __restrict__ in, __bf16* __restrict__ out,
                               int K, int N) {
  __shared__ float tile[32][33];
  int n0 = blockIdx.x * 32, k0 = blockIdx.y * 32;
  int tx = threadIdx.x, ty = threadIdx.y;  // 32 x 8
#pragma unroll
  for (int r = 0; r < 32; r += 8)
    tile[ty + r][tx] = in[(size_t)(k0 + ty + r) * N + n0 + tx];
  __syncthreads();
#pragma unroll
  for (int r = 0; r < 32; r += 8)
    out[(size_t)(n0 + ty + r) * K + k0 + tx] = (__bf16)tile[tx][ty + r];
}

// ---- WMMA 16-bit fragment (ISA 7.12.2): lane l = (row = l&15, hi = l>>4);
// halves 0..7 <- K = k0+8*hi .. +7 ; halves 8..15 <- K = k0+16+8*hi .. +7.
// Both runs are contiguous bf16 -> two b128 loads.
__device__ __forceinline__ v16bf load_frag(const __bf16* __restrict__ rowp,
                                           int k0, int hi) {
  const __bf16* p = rowp + k0 + 8 * hi;
  v8bf lo = *(const v8bf*)(p);
  v8bf hh = *(const v8bf*)(p + 16);
  return __builtin_shufflevector(lo, hh, 0, 1, 2, 3, 4, 5, 6, 7,
                                 8, 9, 10, 11, 12, 13, 14, 15);
}

// ---- GEMM: Out[M,N] = act( A[M,K] @ W[K,N] ),  A bf16 [M,K], WT bf16 [N,K].
// Wave tile 32x64 (8 wmma / K-step); block = 4 waves (2x2) -> 64x128 tile.
// ACT: 0=none 1=tanh 2=sigmoid 3=relu^2
// OM : 0 = f32 store, 1 = f32 accumulate (+=), 2 = bf16 store
// FULLN: all 4 N-subtiles valid (N % 128 == 0) -> branch-free inner loop.
// Requires M % 64 == 0 (true: M = 1536 everywhere), K % 32 == 0.
template <int ACT, int OM, bool FULLN>
__global__ void gemm_k(const __bf16* __restrict__ A,
                       const __bf16* __restrict__ WT,
                       void* __restrict__ OutP, int M, int N, int K) {
  const int wave = threadIdx.x >> 5;
  const int lane = threadIdx.x & 31;
  const int tileM = blockIdx.y * 64 + (wave >> 1) * 32;
  const int tileN = blockIdx.x * 128 + (wave & 1) * 64;
  if (tileM >= M || tileN >= N) return;  // wave-uniform: EXEC stays all-1s
  const int hi = lane >> 4, ml = lane & 15;

  v8f acc[2][4];
#pragma unroll
  for (int mi = 0; mi < 2; ++mi)
#pragma unroll
    for (int j = 0; j < 4; ++j)
      acc[mi][j] = (v8f){0.f, 0.f, 0.f, 0.f, 0.f, 0.f, 0.f, 0.f};

  const __bf16* arow0 = A + (size_t)(tileM + ml) * K;
  const __bf16* arow1 = A + (size_t)(tileM + 16 + ml) * K;
  const __bf16* brow[4];
#pragma unroll
  for (int j = 0; j < 4; ++j)
    brow[j] = WT + (size_t)(tileN + 16 * j + ml) * K;

  for (int k0 = 0; k0 < K; k0 += 32) {
    // speculative prefetch of next K-slab (silently dropped past the end)
    __builtin_prefetch(arow0 + k0 + 32, 0, 3);
    __builtin_prefetch(brow[0] + k0 + 32, 0, 3);
    v16bf a0 = load_frag(arow0, k0, hi);
    v16bf a1 = load_frag(arow1, k0, hi);
#pragma unroll
    for (int j = 0; j < 4; ++j) {
      if (FULLN || tileN + 16 * j < N) {
        v16bf b = load_frag(brow[j], k0, hi);
        acc[0][j] = __builtin_amdgcn_wmma_f32_16x16x32_bf16(
            false, a0, false, b, (short)0, acc[0][j], false, false);
        acc[1][j] = __builtin_amdgcn_wmma_f32_16x16x32_bf16(
            false, a1, false, b, (short)0, acc[1][j], false, false);
      }
    }
  }
  // D layout: VGPR v -> row m = tile + v + 8*hi, col n = tile + (lane&15)
#pragma unroll
  for (int mi = 0; mi < 2; ++mi) {
#pragma unroll
    for (int j = 0; j < 4; ++j) {
      if (!(FULLN || tileN + 16 * j < N)) continue;
      int n = tileN + 16 * j + ml;
#pragma unroll
      for (int v = 0; v < 8; ++v) {
        int m = tileM + 16 * mi + v + 8 * hi;
        float val = acc[mi][j][v];
        if (ACT == 1)      val = tanhf(val);
        else if (ACT == 2) val = 1.f / (1.f + expf(-val));
        else if (ACT == 3) { val = fmaxf(val, 0.f); val *= val; }
        size_t idx = (size_t)m * N + n;
        if (OM == 2)      ((__bf16*)OutP)[idx] = (__bf16)val;
        else if (OM == 1) ((float*)OutP)[idx] += val;
        else              ((float*)OutP)[idx] = val;
      }
    }
  }
}

// ---- Embedding gather: X[b,t,:] = table[token][:] --------------------------
__global__ void embed_kernel(const int* __restrict__ text,
                             const int* __restrict__ audio,
                             const float* __restrict__ temb,
                             const float* __restrict__ aemb,
                             float* __restrict__ X, int Tt, int Ta, int C) {
  int row = blockIdx.x;  // b*T + t
  int T = Tt + Ta;
  int b = row / T, t = row % T;
  const float* src = (t < Tt) ? (temb + (size_t)text[b * Tt + t] * C)
                              : (aemb + (size_t)audio[b * Ta + (t - Tt)] * C);
  float* dst = X + (size_t)row * C;
  for (int c = threadIdx.x; c < C; c += blockDim.x) dst[c] = src[c];
}

// ---- LayerNorm over last dim (block/row); optional extra bf16 output -------
__global__ void ln_kernel(const float* __restrict__ x, const float* __restrict__ w,
                          const float* __restrict__ b, float* __restrict__ out,
                          __bf16* __restrict__ outb, int C) {
  int row = blockIdx.x;
  const float* xr = x + (size_t)row * C;
  __shared__ float red[256];
  float s = 0.f;
  for (int c = threadIdx.x; c < C; c += 256) s += xr[c];
  red[threadIdx.x] = s; __syncthreads();
  for (int st = 128; st > 0; st >>= 1) {
    if (threadIdx.x < st) red[threadIdx.x] += red[threadIdx.x + st];
    __syncthreads();
  }
  float mean = red[0] / C;
  __syncthreads();
  float v = 0.f;
  for (int c = threadIdx.x; c < C; c += 256) { float d = xr[c] - mean; v += d * d; }
  red[threadIdx.x] = v; __syncthreads();
  for (int st = 128; st > 0; st >>= 1) {
    if (threadIdx.x < st) red[threadIdx.x] += red[threadIdx.x + st];
    __syncthreads();
  }
  float rstd = rsqrtf(red[0] / C + 1e-5f);
  float* o = out + (size_t)row * C;
  for (int c = threadIdx.x; c < C; c += 256) {
    float val = (xr[c] - mean) * rstd * w[c] + b[c];
    o[c] = val;
    if (outb) outb[(size_t)row * C + c] = (__bf16)val;
  }
}

// ---- token-shift + 6 mixes, bf16 outputs (feed WMMA GEMMs) -----------------
__global__ void shift_mix6(const float* __restrict__ xn,
                           const float* m0, const float* m1, const float* m2,
                           const float* m3, const float* m4, const float* m5,
                           __bf16* o0, __bf16* o1, __bf16* o2,
                           __bf16* o3, __bf16* o4, __bf16* o5, int T, int C) {
  int row = blockIdx.x;
  int t = row % T;
  size_t base = (size_t)row * C;
  for (int c = threadIdx.x; c < C; c += blockDim.x) {
    float cur = xn[base + c];
    float prev = (t == 0) ? 0.f : xn[base - C + c];
    float xx = prev - cur;
    o0[base + c] = (__bf16)(cur + xx * m0[c]);
    o1[base + c] = (__bf16)(cur + xx * m1[c]);
    o2[base + c] = (__bf16)(cur + xx * m2[c]);
    o3[base + c] = (__bf16)(cur + xx * m3[c]);
    o4[base + c] = (__bf16)(cur + xx * m4[c]);
    o5[base + c] = (__bf16)(cur + xx * m5[c]);
  }
}

__global__ void shift_mix1(const float* __restrict__ xn, const float* m0,
                           __bf16* o0, int T, int C) {
  int row = blockIdx.x;
  int t = row % T;
  size_t base = (size_t)row * C;
  for (int c = threadIdx.x; c < C; c += blockDim.x) {
    float cur = xn[base + c];
    float prev = (t == 0) ? 0.f : xn[base - C + c];
    o0[base + c] = (__bf16)(cur + (prev - cur) * m0[c]);
  }
}

// ---- tmix elementwise prep: decay, v_first blend, kk-norm, ab/bb, k --------
__global__ void tmix_prep(float* __restrict__ Wbuf, float* __restrict__ Abuf,
                          const float* __restrict__ VGbuf, float* __restrict__ Vbuf,
                          float* __restrict__ VFirst, float* __restrict__ Kbuf,
                          float* __restrict__ ABbuf, float* __restrict__ BBbuf,
                          const float* __restrict__ w0, const float* __restrict__ a0,
                          const float* __restrict__ v0, const float* __restrict__ k_k,
                          const float* __restrict__ k_a, int layer) {
  const int C = C_DIM;
  int row = blockIdx.x;
  size_t base = (size_t)row * C;
  __shared__ float kkS[C_DIM];
  __shared__ float aS[C_DIM];
  __shared__ float nrm[HEADS];
  for (int c = threadIdx.x; c < C; c += blockDim.x) {
    float a = 1.f / (1.f + expf(-(a0[c] + Abuf[base + c])));
    aS[c] = a;
    Abuf[base + c] = a;
    float wv = w0[c] + Wbuf[base + c];
    float w = -log1pf(expf(-wv)) - 0.5f;   // -softplus(-wv) - 0.5
    Wbuf[base + c] = expf(-expf(w));       // per-step decay
    float vcur = Vbuf[base + c];
    if (layer == 0) {
      VFirst[base + c] = vcur;
    } else {
      float g = 1.f / (1.f + expf(-(v0[c] + VGbuf[base + c])));
      vcur = vcur + (VFirst[base + c] - vcur) * g;
      Vbuf[base + c] = vcur;
    }
    float kv = Kbuf[base + c];
    kkS[c] = kv * k_k[c];
    Kbuf[base + c] = kv * (1.f + (a - 1.f) * k_a[c]);
  }
  __syncthreads();
  if (threadIdx.x < HEADS) {
    float s = 0.f;
    for (int j = 0; j < HEAD_N; ++j) {
      float t = kkS[threadIdx.x * HEAD_N + j];
      s += t * t;
    }
    nrm[threadIdx.x] = fmaxf(sqrtf(s), 1e-12f);
  }
  __syncthreads();
  for (int c = threadIdx.x; c < C; c += blockDim.x) {
    float kkn = kkS[c] / nrm[c >> 6];
    ABbuf[base + c] = -kkn;
    BBbuf[base + c] = kkn * aS[c];
  }
}

// ---- RWKV7 recurrence: one block per (b,h); lane i owns state row i --------
__global__ void rwkv_scan(const float* __restrict__ r, const float* __restrict__ wd,
                          const float* __restrict__ k, const float* __restrict__ v,
                          const float* __restrict__ ab, const float* __restrict__ bb,
                          float* __restrict__ o, int B, int T, int H) {
  const int C = H * HEAD_N;
  int bh = blockIdx.x;
  int b = bh / H, h = bh % H;
  int i = threadIdx.x;  // 0..63
  float S[HEAD_N];
#pragma unroll
  for (int j = 0; j < HEAD_N; ++j) S[j] = 0.f;
  __shared__ float sr[HEAD_N], sw[HEAD_N], sk[HEAD_N], sv[HEAD_N],
                   sab[HEAD_N], sbb[HEAD_N];
  for (int t = 0; t < T; ++t) {
    size_t base = ((size_t)(b * T + t)) * C + h * HEAD_N;
    sr[i] = r[base + i];  sw[i] = wd[base + i];  sk[i] = k[base + i];
    sv[i] = v[base + i];  sab[i] = ab[base + i]; sbb[i] = bb[base + i];
    __syncthreads();
    float dot_a = 0.f;
#pragma unroll 8
    for (int j = 0; j < HEAD_N; ++j) dot_a += S[j] * sab[j];
    float vi = sv[i];
    float y = 0.f;
#pragma unroll 8
    for (int j = 0; j < HEAD_N; ++j) {
      float s = S[j] * sw[j] + dot_a * sbb[j] + vi * sk[j];
      S[j] = s;
      y += s * sr[j];
    }
    o[base + i] = y;
    __syncthreads();
  }
}

// ---- GroupNorm(H, eps=6.4e-4) + rk*v + gate -> bf16 (feeds Wo GEMM) --------
__global__ void gn_rk_gate(const float* __restrict__ O, const float* __restrict__ R,
                           const float* __restrict__ K, const float* __restrict__ V,
                           const float* __restrict__ G, const float* __restrict__ r_k,
                           const float* __restrict__ gw, const float* __restrict__ gb,
                           __bf16* __restrict__ Ob16) {
  const int C = C_DIM;
  int row = blockIdx.x;
  size_t base = (size_t)row * C;
  __shared__ float mean[HEADS], rstd[HEADS], rks[HEADS];
  if (threadIdx.x < HEADS) {
    int h = threadIdx.x;
    float m = 0.f, s2 = 0.f, rk = 0.f;
    for (int n = 0; n < HEAD_N; ++n) {
      size_t idx = base + h * HEAD_N + n;
      float o = O[idx];
      m += o; s2 += o * o;
      rk += R[idx] * K[idx] * r_k[h * HEAD_N + n];
    }
    m /= (float)HEAD_N;
    s2 = s2 / (float)HEAD_N - m * m;
    mean[h] = m; rstd[h] = rsqrtf(s2 + 0.00064f); rks[h] = rk;
  }
  __syncthreads();
  for (int c = threadIdx.x; c < C; c += blockDim.x) {
    int h = c >> 6;
    float o = (O[base + c] - mean[h]) * rstd[h] * gw[c] + gb[c];
    o += rks[h] * V[base + c];
    Ob16[base + c] = (__bf16)(o * G[base + c]);
  }
}

// ---------------------------------------------------------------------------
extern "C" void kernel_launch(void* const* d_in, const int* in_sizes, int n_in,
                              void* d_out, int out_size, void* d_ws, size_t ws_size,
                              hipStream_t stream) {
  const int B = 2, Tt = 128, Ta = 640, T = Tt + Ta, BT = B * T;
  const int C = C_DIM, H = HEADS, NB = 2, VA = 8192;
  (void)in_sizes; (void)n_in; (void)out_size; (void)ws_size;

  const int* text  = (const int*)d_in[0];
  const int* audio = (const int*)d_in[2];
  auto F = [&](int i) { return (const float*)d_in[i]; };
  const float* audio_embed = F(3);
  const float* head_w      = F(70);
  const float* ln_in_b     = F(71);
  const float* ln_in_w     = F(72);
  const float* ln_out_b    = F(73);
  const float* ln_out_w    = F(74);
  const float* text_embed  = F(75);
  // att leaves: 0 Wk 1 Wo 2 Wr 3 Wv 4 a0 5 a1 6 a2 7 g1 8 g2 9 k_a 10 k_k
  // 11 ln_x_b 12 ln_x_w 13 r_k 14 v0 15 v1 16 v2 17 w0 18 w1 19 w2
  // 20 x_a 21 x_g 22 x_k 23 x_r 24 x_v 25 x_w ; ffn: 26 Wk 27 Wv 28 x_k ;
  // 29 ln1_b 30 ln1_w 31 ln2_b 32 ln2_w
  auto BP = [&](int blk, int leaf) { return (const float*)d_in[4 + 33 * blk + leaf]; };

  // ---- workspace carve-out ----
  char* cur = (char*)d_ws;
  auto allocF = [&](size_t n) {
    float* r = (float*)cur; cur += ((n * 4 + 255) & ~(size_t)255); return r;
  };
  auto allocB = [&](size_t n) {
    __bf16* r = (__bf16*)cur; cur += ((n * 2 + 255) & ~(size_t)255); return r;
  };
  const size_t f = (size_t)BT * C;
  float* X   = allocF(f);
  float* VF  = allocF(f);
  float* XN  = allocF(f);
  float* Rb  = allocF(f);
  float* Kb  = allocF(f);
  float* Vb  = allocF(f);
  float* Wb  = allocF(f);
  float* Ab  = allocF(f);
  float* Gb  = allocF(f);
  float* ABb = allocF(f);
  float* BBb = allocF(f);
  float* Ob  = allocF(f);
  float* VGb = allocF(f);
  __bf16* M0 = allocB(f);
  __bf16* M1 = allocB(f);
  __bf16* M2 = allocB(f);
  __bf16* M3 = allocB(f);
  __bf16* M4 = allocB(f);
  __bf16* M5 = allocB(f);
  __bf16* FFNH = allocB((size_t)BT * 4 * C);
  __bf16* T1b  = allocB((size_t)BT * 128);
  __bf16* Obf  = allocB(f);
  __bf16* XNb  = allocB(f);
  // bf16 weight arena (recast per block; head GEMM reuses it afterwards)
  __bf16* WA_Wk = allocB((size_t)C * C);
  __bf16* WA_Wo = allocB((size_t)C * C);
  __bf16* WA_Wr = allocB((size_t)C * C);
  __bf16* WA_Wv = allocB((size_t)C * C);
  __bf16* WA_w1 = allocB((size_t)64 * C);
  __bf16* WA_w2 = allocB((size_t)64 * C);
  __bf16* WA_a1 = allocB((size_t)64 * C);
  __bf16* WA_a2 = allocB((size_t)64 * C);
  __bf16* WA_v1 = allocB((size_t)32 * C);
  __bf16* WA_v2 = allocB((size_t)32 * C);
  __bf16* WA_g1 = allocB((size_t)128 * C);
  __bf16* WA_g2 = allocB((size_t)128 * C);
  __bf16* WA_fk = allocB((size_t)4 * C * C);
  __bf16* WA_fv = allocB((size_t)4 * C * C);

  auto castT = [&](const float* src, __bf16* dst, int K_, int N_) {
    cast_transpose<<<dim3(N_ / 32, K_ / 32), dim3(32, 8), 0, stream>>>(src, dst, K_, N_);
  };
  auto gemm = [&](const __bf16* Ap, const __bf16* WTp, void* Op,
                  int M, int N, int K, int act, int om) {
    dim3 grid((N + 127) / 128, (M + 63) / 64);
    if ((N & 127) == 0) {  // branch-free inner loop
      if (act == 0 && om == 0)
        gemm_k<0, 0, true><<<grid, 128, 0, stream>>>(Ap, WTp, Op, M, N, K);
      else if (act == 0 && om == 1)
        gemm_k<0, 1, true><<<grid, 128, 0, stream>>>(Ap, WTp, Op, M, N, K);
      else if (act == 2 && om == 2)
        gemm_k<2, 2, true><<<grid, 128, 0, stream>>>(Ap, WTp, Op, M, N, K);
      else  // act == 3, om == 2
        gemm_k<3, 2, true><<<grid, 128, 0, stream>>>(Ap, WTp, Op, M, N, K);
    } else {               // skinny LoRA shapes (N = 32 / 64)
      if (act == 1)
        gemm_k<1, 2, false><<<grid, 128, 0, stream>>>(Ap, WTp, Op, M, N, K);
      else
        gemm_k<0, 2, false><<<grid, 128, 0, stream>>>(Ap, WTp, Op, M, N, K);
    }
  };

  // ---- embed + input LN ----
  embed_kernel<<<BT, 256, 0, stream>>>(text, audio, text_embed, audio_embed, X, Tt, Ta, C);
  ln_kernel<<<BT, 256, 0, stream>>>(X, ln_in_w, ln_in_b, X, (__bf16*)nullptr, C);

  for (int blk = 0; blk < NB; ++blk) {
    // recast this block's weights into the bf16 [N,K] arena
    castT(BP(blk, 0),  WA_Wk, C, C);
    castT(BP(blk, 1),  WA_Wo, C, C);
    castT(BP(blk, 2),  WA_Wr, C, C);
    castT(BP(blk, 3),  WA_Wv, C, C);
    castT(BP(blk, 18), WA_w1, C, 64);
    castT(BP(blk, 19), WA_w2, 64, C);
    castT(BP(blk, 5),  WA_a1, C, 64);
    castT(BP(blk, 6),  WA_a2, 64, C);
    castT(BP(blk, 15), WA_v1, C, 32);
    castT(BP(blk, 16), WA_v2, 32, C);
    castT(BP(blk, 7),  WA_g1, C, 128);
    castT(BP(blk, 8),  WA_g2, 128, C);
    castT(BP(blk, 26), WA_fk, C, 4 * C);
    castT(BP(blk, 27), WA_fv, 4 * C, C);

    // ---- time-mix ----
    ln_kernel<<<BT, 256, 0, stream>>>(X, BP(blk, 30), BP(blk, 29), XN,
                                      (__bf16*)nullptr, C);
    shift_mix6<<<BT, 256, 0, stream>>>(XN,
        BP(blk, 23), BP(blk, 25), BP(blk, 22), BP(blk, 24), BP(blk, 20), BP(blk, 21),
        M0, M1, M2, M3, M4, M5, T, C);            // xr xw xk xv xa xg
    gemm(M0, WA_Wr, Rb, BT, C, C, 0, 0);          // r = xr @ Wr
    gemm(M2, WA_Wk, Kb, BT, C, C, 0, 0);          // k = xk @ Wk
    gemm(M3, WA_Wv, Vb, BT, C, C, 0, 0);          // v = xv @ Wv
    gemm(M1, WA_w1, T1b, BT, 64, C, 1, 2);        // tanh(xw @ w1) -> bf16
    gemm(T1b, WA_w2, Wb, BT, C, 64, 0, 0);        // @ w2
    gemm(M4, WA_a1, T1b, BT, 64, C, 0, 2);        // xa @ a1 -> bf16
    gemm(T1b, WA_a2, Ab, BT, C, 64, 0, 0);        // @ a2
    if (blk > 0) {
      gemm(M3, WA_v1, T1b, BT, 32, C, 0, 2);      // xv @ v1 -> bf16
      gemm(T1b, WA_v2, VGb, BT, C, 32, 0, 0);     // @ v2
    }
    gemm(M5, WA_g1, T1b, BT, 128, C, 2, 2);       // sigmoid(xg @ g1) -> bf16
    gemm(T1b, WA_g2, Gb, BT, C, 128, 0, 0);       // @ g2
    tmix_prep<<<BT, 256, 0, stream>>>(Wb, Ab, VGb, Vb, VF, Kb, ABb, BBb,
        BP(blk, 17), BP(blk, 4), BP(blk, 14), BP(blk, 10), BP(blk, 9), blk);
    rwkv_scan<<<B * H, HEAD_N, 0, stream>>>(Rb, Wb, Kb, Vb, ABb, BBb, Ob, B, T, H);
    gn_rk_gate<<<BT, 256, 0, stream>>>(Ob, Rb, Kb, Vb, Gb,
        BP(blk, 13), BP(blk, 12), BP(blk, 11), Obf);
    gemm(Obf, WA_Wo, X, BT, C, C, 0, 1);          // x += (o*g) @ Wo

    // ---- channel-mix ----
    ln_kernel<<<BT, 256, 0, stream>>>(X, BP(blk, 32), BP(blk, 31), XN,
                                      (__bf16*)nullptr, C);
    shift_mix1<<<BT, 256, 0, stream>>>(XN, BP(blk, 28), M0, T, C);
    gemm(M0, WA_fk, FFNH, BT, 4 * C, C, 3, 2);    // relu(kx@Wk)^2 -> bf16
    gemm(FFNH, WA_fv, X, BT, C, 4 * C, 0, 1);     // x += h @ Wv
  }

  // ---- output head (reuse weight arena for head^T) ----
  __bf16* headT = WA_Wk;  // 8192x1024 bf16 = 16 MB fits in arena (26 MB)
  castT(head_w, headT, C, VA);
  ln_kernel<<<BT, 256, 0, stream>>>(X, ln_out_w, ln_out_b, XN, XNb, C);
  gemm(XNb, headT, (float*)d_out, BT, VA, C, 0, 0);
}
